// MultiViewFullBatchGAE_36764920054544
// MI455X (gfx1250) — matrix operand order repaired and checked by reading, not compiled, not run on hardware
//
#include <hip/hip_runtime.h>
#include <hip/hip_bf16.h>
#include <math.h>

#define N_NODES 200000
#define NNZ_CNT 6400000
#define E_CNT   500000
#define HID     256
#define LAT     128
#define DHID    64

typedef __attribute__((ext_vector_type(16))) _Float16 v16h;
typedef __attribute__((ext_vector_type(8)))  _Float16 v8h;
typedef __attribute__((ext_vector_type(8)))  float    v8f;

// Fragment geometry (ISA 7.12.2, wave32, 16-bit operands, 16x16x32):
//  A[m,k]: lane m = lane&15; half = lane>>4; V0..3 hold K = half*8 + 0..7 (contiguous),
//          V4..7 hold K = 16 + half*8 + 0..7 (contiguous)  -> 2x b128 from row-major f16.
//  B[k,n]: lane n = lane&15; V0..3 hold K = half*8 + 0..7, V4..7 hold K = 16 + half*8 + 0..7
//          -> 2x b128 from K-contiguous (transposed) f16 weights.

__device__ __forceinline__ v16h cat8(v8h lo, v8h hi) {
  return __builtin_shufflevector(lo, hi, 0, 1, 2, 3, 4, 5, 6, 7,
                                 8, 9, 10, 11, 12, 13, 14, 15);
}

// ---------------- COO SpMM: S[row,:] += val * X[col,:] ----------------
template<int F>
__global__ __launch_bounds__(256)
void spmm_coo(const int* __restrict__ rows, const int* __restrict__ cols,
              const float* __restrict__ vals, const float* __restrict__ X,
              float* __restrict__ S, int nnz)
{
  const int f = threadIdx.x;
  const long long e = (long long)blockIdx.x * blockDim.y + threadIdx.y;
  if (e >= nnz) return;
  if (f >= F) return;
  const int r = rows[e];
  const int c = cols[e];
  const float v = vals[e];
  const float contrib = v * X[(size_t)c * F + f];
  __hip_atomic_fetch_add(&S[(size_t)r * F + f], contrib,
                         __ATOMIC_RELAXED, __HIP_MEMORY_SCOPE_AGENT);
}

// ------- activations f32 [M,KIN] -> f16 [M,KPAD], zero padded ---------
__global__ __launch_bounds__(256)
void cvt_pad_f16(const float* __restrict__ A, _Float16* __restrict__ O,
                 int M, int KIN, int KPAD)
{
  const long long g = (long long)blockIdx.x * blockDim.x + threadIdx.x;
  const int gper = KPAD >> 3;                 // 8-half groups per row
  const long long total = (long long)M * gper;
  if (g >= total) return;
  const int m  = (int)(g / gper);
  const int k0 = (int)(g - (long long)m * gper) * 8;
  const float* src = A + (size_t)m * KIN;
  v8h o;
#pragma unroll
  for (int j = 0; j < 8; ++j)
    o[j] = (k0 + j < KIN) ? (_Float16)src[k0 + j] : (_Float16)0.f;
  *reinterpret_cast<v8h*>(O + (size_t)m * KPAD + k0) = o;
}

// ------ weights f32 [KIN,Ncol] -> f16 transposed [Ncol,KPAD] ----------
__global__ __launch_bounds__(256)
void xpose_w_f16(const float* __restrict__ W, _Float16* __restrict__ WT,
                 int KIN, int Ncol, int KPAD)
{
  const int tid = blockIdx.x * blockDim.x + threadIdx.x;
  if (tid >= Ncol * KPAD) return;
  const int n = tid / KPAD;
  const int k = tid - n * KPAD;
  WT[tid] = (k < KIN) ? (_Float16)W[(size_t)k * Ncol + n] : (_Float16)0.f;
}

// ---- C = act(A16[M,KPAD] @ WT[Ncol,KPAD]^T + bias), 16x64 per wave ---
template<int KPAD, bool RELU>
__global__ __launch_bounds__(256)
void gemm_bias_wmma(const _Float16* __restrict__ A16,
                    const _Float16* __restrict__ WT,
                    const float* __restrict__ bias, float* __restrict__ C,
                    int M, int Ncol)
{
  const int lane = threadIdx.x & 31;
  const int wave = threadIdx.x >> 5;
  const int ngrp = Ncol >> 6;                 // 64-wide column groups
  const int w    = blockIdx.x * 8 + wave;     // wave-uniform
  const int tileM = w / ngrp;
  const int cg    = w - tileM * ngrp;
  if (tileM * 16 >= M) return;                // uniform exit: EXEC stays full
  const int hf = lane >> 4;
  const int nl = lane & 15;
  const _Float16* arow = A16 + (size_t)(tileM * 16 + nl) * KPAD + hf * 8;

  v8f acc[4] = {};
#pragma unroll
  for (int kb = 0; kb < KPAD; kb += 32) {
    const v8h alo = *reinterpret_cast<const v8h*>(arow + kb);
    const v8h ahi = *reinterpret_cast<const v8h*>(arow + kb + 16);
    const v16h a = cat8(alo, ahi);
#pragma unroll
    for (int t = 0; t < 4; ++t) {
      const int ncol = (cg * 4 + t) * 16 + nl;
      const _Float16* bp = WT + (size_t)ncol * KPAD + kb + hf * 8;
      const v8h blo = *reinterpret_cast<const v8h*>(bp);
      const v8h bhi = *reinterpret_cast<const v8h*>(bp + 16);
      const v16h b = cat8(blo, bhi);
      acc[t] = __builtin_amdgcn_wmma_f32_16x16x32_f16(false, a, false, b,
                                                      (short)0, acc[t], false, false);
    }
  }
#pragma unroll
  for (int t = 0; t < 4; ++t) {
    const int ncol = (cg * 4 + t) * 16 + nl;
    const float bc = bias[ncol];
#pragma unroll
    for (int r = 0; r < 8; ++r) {
      float v = acc[t][r] + bc;
      if (RELU) v = fmaxf(v, 0.f);
      C[(size_t)(tileM * 16 + r + 8 * hf) * Ncol + ncol] = v;
    }
  }
}

// ---------------- softmax gate over 3 views, fuse latents -------------
__global__ __launch_bounds__(256)
void gate_fuse(const float* __restrict__ zp, const float* __restrict__ zf,
               const float* __restrict__ zn,
               const float* __restrict__ Wgp, const float* __restrict__ bgp,
               const float* __restrict__ Wgf, const float* __restrict__ bgf,
               const float* __restrict__ Wgn, const float* __restrict__ bgn,
               float* __restrict__ z, int n)
{
  const int i = blockIdx.x * blockDim.x + threadIdx.x;
  if (i >= n) return;
  const float* p = zp + (size_t)i * LAT;
  const float* f = zf + (size_t)i * LAT;
  const float* q = zn + (size_t)i * LAT;
  float sp = bgp[0], sf = bgf[0], sn = bgn[0];
  for (int j = 0; j < LAT; ++j) {
    sp += p[j] * Wgp[j];
    sf += f[j] * Wgf[j];
    sn += q[j] * Wgn[j];
  }
  const float mx = fmaxf(sp, fmaxf(sf, sn));
  const float ep = __expf(sp - mx), ef = __expf(sf - mx), en = __expf(sn - mx);
  const float inv = 1.f / (ep + ef + en);
  const float ap = ep * inv, af = ef * inv, an = en * inv;
  float* zo = z + (size_t)i * LAT;
  for (int j = 0; j < LAT; ++j)
    zo[j] = ap * p[j] + af * f[j] + an * q[j];
}

// ----- edge decoder: feats[E,512] -> relu(@Wd1+bd1) -> @Wd2+bd2 -------
__global__ __launch_bounds__(64)
void edge_decoder_wmma(const float* __restrict__ z,
                       const int* __restrict__ esrc, const int* __restrict__ edst,
                       const _Float16* __restrict__ Wd1T,   // [64, 512] f16
                       const float* __restrict__ bd1,
                       const float* __restrict__ Wd2, const float* __restrict__ bd2,
                       float* __restrict__ out, int ne)
{
  __shared__ __align__(16) _Float16 feats[2][16][4 * LAT];  // 32 KB
  const int lane = threadIdx.x & 31;
  const int wave = threadIdx.x >> 5;
  const long long g = (long long)blockIdx.x * 2 + wave;     // 16-edge group
  const bool active = (g * 16 < ne);                        // wave-uniform
  _Float16 (*fw)[4 * LAT] = feats[wave];

  if (active) {
    for (int m = 0; m < 16; ++m) {
      const long long e = g * 16 + m;
      const float* zs = z + (size_t)esrc[e] * LAT;
      const float* zd = z + (size_t)edst[e] * LAT;
      for (int j = lane; j < LAT; j += 32) {
        const float a = zs[j], b = zd[j];
        fw[m][j]           = (_Float16)a;
        fw[m][LAT + j]     = (_Float16)b;
        fw[m][2 * LAT + j] = (_Float16)(a * b);
        fw[m][3 * LAT + j] = (_Float16)fabsf(a - b);
      }
    }
  }
  __syncthreads();
  if (!active) return;

  const int hf = lane >> 4, nl = lane & 15;
  const _Float16* arow = &fw[nl][hf * 8];
  v8f acc[4] = {};
#pragma unroll
  for (int kb = 0; kb < 4 * LAT; kb += 32) {
    const v8h alo = *reinterpret_cast<const v8h*>(arow + kb);       // ds_load_b128
    const v8h ahi = *reinterpret_cast<const v8h*>(arow + kb + 16);
    const v16h a = cat8(alo, ahi);
#pragma unroll
    for (int t = 0; t < 4; ++t) {
      const _Float16* bp = Wd1T + (size_t)(t * 16 + nl) * (4 * LAT) + kb + hf * 8;
      const v8h blo = *reinterpret_cast<const v8h*>(bp);
      const v8h bhi = *reinterpret_cast<const v8h*>(bp + 16);
      const v16h b = cat8(blo, bhi);
      acc[t] = __builtin_amdgcn_wmma_f32_16x16x32_f16(false, a, false, b,
                                                      (short)0, acc[t], false, false);
    }
  }
  // layer 2: relu(h + bd1) . Wd2 over 64 cols, reduced across 16 lanes
  float part[8] = {0.f, 0.f, 0.f, 0.f, 0.f, 0.f, 0.f, 0.f};
#pragma unroll
  for (int t = 0; t < 4; ++t) {
    const int col = t * 16 + nl;
    const float bb = bd1[col], w2 = Wd2[col];
#pragma unroll
    for (int r = 0; r < 8; ++r)
      part[r] += fmaxf(acc[t][r] + bb, 0.f) * w2;
  }
#pragma unroll
  for (int off = 1; off < 16; off <<= 1) {
#pragma unroll
    for (int r = 0; r < 8; ++r)
      part[r] += __shfl_xor(part[r], off, 32);
  }
  if (nl == 0) {
    const float b2 = bd2[0];
#pragma unroll
    for (int r = 0; r < 8; ++r)
      out[g * 16 + r + 8 * hf] = part[r] + b2;
  }
}

extern "C" void kernel_launch(void* const* d_in, const int* in_sizes, int n_in,
                              void* d_out, int out_size, void* d_ws, size_t ws_size,
                              hipStream_t stream)
{
  (void)in_sizes; (void)n_in; (void)out_size; (void)ws_size;

  const float* x[3]  = {(const float*)d_in[0], (const float*)d_in[1], (const float*)d_in[2]};
  const int*   rows  = (const int*)d_in[3];
  const int*   cols  = (const int*)d_in[4];
  const float* vals  = (const float*)d_in[5];
  const int*   esrc  = (const int*)d_in[6];
  const int*   edst  = (const int*)d_in[7];
  const float* W1[3] = {(const float*)d_in[8],  (const float*)d_in[12], (const float*)d_in[16]};
  const float* b1[3] = {(const float*)d_in[9],  (const float*)d_in[13], (const float*)d_in[17]};
  const float* W2[3] = {(const float*)d_in[10], (const float*)d_in[14], (const float*)d_in[18]};
  const float* b2[3] = {(const float*)d_in[11], (const float*)d_in[15], (const float*)d_in[19]};
  const float* Wg[3] = {(const float*)d_in[20], (const float*)d_in[22], (const float*)d_in[24]};
  const float* bg[3] = {(const float*)d_in[21], (const float*)d_in[23], (const float*)d_in[25]};
  const float* Wd1   = (const float*)d_in[26];
  const float* bd1   = (const float*)d_in[27];
  const float* Wd2   = (const float*)d_in[28];
  const float* bd2   = (const float*)d_in[29];
  float* out = (float*)d_out;

  const int F[3] = {62, 37, 37};

  // workspace carve-up:
  // s1 f32 N*64 | h f32 N*256 | s2 f32 N*256 | zv0..2,zfu f32 N*128 each |
  // a16 f16 N*256 | wt1 f16 256*64 | wt2 f16 128*256 | wd1t f16 64*512
  float* s1  = (float*)d_ws;
  float* h   = s1  + (size_t)N_NODES * 64;
  float* s2  = h   + (size_t)N_NODES * HID;
  float* zv0 = s2  + (size_t)N_NODES * HID;
  float* zv1 = zv0 + (size_t)N_NODES * LAT;
  float* zv2 = zv1 + (size_t)N_NODES * LAT;
  float* zfu = zv2 + (size_t)N_NODES * LAT;
  _Float16* a16  = (_Float16*)(zfu + (size_t)N_NODES * LAT);
  _Float16* wt1  = a16 + (size_t)N_NODES * HID;      // [256, 64]
  _Float16* wt2  = wt1 + (size_t)HID * 64;           // [128, 256]
  _Float16* wd1t = wt2 + (size_t)LAT * HID;          // [64, 512]
  float* zview[3] = {zv0, zv1, zv2};

  for (int v = 0; v < 3; ++v) {
    hipMemsetAsync(s1, 0, (size_t)N_NODES * F[v] * sizeof(float), stream);
    {
      dim3 blk(64, 4);
      const int grid = (NNZ_CNT + 3) / 4;
      if (v == 0)
        spmm_coo<62><<<grid, blk, 0, stream>>>(rows, cols, vals, x[v], s1, NNZ_CNT);
      else
        spmm_coo<37><<<grid, blk, 0, stream>>>(rows, cols, vals, x[v], s1, NNZ_CNT);
    }
    {
      const long long tot = (long long)N_NODES * (64 / 8);
      cvt_pad_f16<<<(int)((tot + 255) / 256), 256, 0, stream>>>(s1, a16, N_NODES, F[v], 64);
      xpose_w_f16<<<(HID * 64 + 255) / 256, 256, 0, stream>>>(W1[v], wt1, F[v], HID, 64);
      const int waves = (N_NODES / 16) * (HID / 64);
      gemm_bias_wmma<64, true><<<(waves + 7) / 8, 256, 0, stream>>>(
          a16, wt1, b1[v], h, N_NODES, HID);
    }
    hipMemsetAsync(s2, 0, (size_t)N_NODES * HID * sizeof(float), stream);
    {
      dim3 blk(256, 1);
      spmm_coo<256><<<NNZ_CNT, blk, 0, stream>>>(rows, cols, vals, h, s2, NNZ_CNT);
    }
    {
      const long long tot = (long long)N_NODES * (HID / 8);
      cvt_pad_f16<<<(int)((tot + 255) / 256), 256, 0, stream>>>(s2, a16, N_NODES, HID, HID);
      xpose_w_f16<<<(LAT * HID + 255) / 256, 256, 0, stream>>>(W2[v], wt2, HID, LAT, HID);
      const int waves = (N_NODES / 16) * (LAT / 64);
      gemm_bias_wmma<256, false><<<(waves + 7) / 8, 256, 0, stream>>>(
          a16, wt2, b2[v], zview[v], N_NODES, LAT);
    }
  }

  gate_fuse<<<(N_NODES + 255) / 256, 256, 0, stream>>>(
      zv0, zv1, zv2, Wg[0], bg[0], Wg[1], bg[1], Wg[2], bg[2], zfu, N_NODES);

  {
    xpose_w_f16<<<(DHID * 512 + 255) / 256, 256, 0, stream>>>(Wd1, wd1t, 512, DHID, 512);
    const int groups = (E_CNT + 15) / 16;
    const int blocks = (groups + 1) / 2;
    edge_decoder_wmma<<<blocks, 64, 0, stream>>>(
        zfu, esrc, edst, wd1t, bd1, Wd2, bd2, out, E_CNT);
  }
}
